// CRFDecoder_17308718203636
// MI455X (gfx1250) — matrix operand order repaired and checked by reading, not compiled, not run on hardware
//
#include <hip/hip_runtime.h>
#include <hip/hip_bf16.h>

// ---------------- problem constants (from reference) ----------------
#define D_IN   768
#define HID    384          // D/2
#define NLAB   9
#define NL     11           // CRF label count (with START/END)
#define LPAD   16           // padded label stride for logits/pointers
#define BATCH  64
#define SEQ    512
#define MROWS  (BATCH*SEQ)  // 32768 rows, divisible by 16
#define KT1    (D_IN/32)    // 24 k-steps for GEMM1
#define NT1    (HID/16)     // 24 n-tiles for GEMM1
#define KT2    (HID/32)     // 12 k-steps for GEMM2
#define PAD_VAL  (-1000.0f)
#define INIT_VAL (-100.0f)

// ---------------- ws layout (bytes) ----------------
#define WS_W1_OFF  0
#define WS_W1_SIZE (NT1*KT1*32*8*4)          // 589824  (packed bf16 W1 fragments)
#define WS_W2_OFF  (WS_W1_OFF + WS_W1_SIZE)
#define WS_W2_SIZE (KT2*32*8*4)              // 12288   (packed bf16 W2 fragments)
#define WS_LG_OFF  (WS_W2_OFF + WS_W2_SIZE)  // 602112
#define WS_LG_SIZE (MROWS*LPAD*4)            // 2 MB    (padded logits, f32)
#define WS_PTR_OFF (WS_LG_OFF + WS_LG_SIZE)  // 2699264
#define WS_PTR_SIZE (MROWS*LPAD)             // 512 KB  (viterbi back-pointers, u8)

typedef __attribute__((ext_vector_type(16))) __bf16       v16bf;
typedef __attribute__((ext_vector_type(8)))  float        v8f;
typedef __attribute__((ext_vector_type(4)))  float        v4f;
typedef __attribute__((ext_vector_type(4)))  unsigned int v4u;

union Frag16 {                 // one 16-bit WMMA A/B operand fragment
    v16bf        v;
    unsigned int u[8];
    v4u          q[2];
};

// exact RNE f32 -> bf16 (used in cold pack kernels / epilogue)
__device__ __forceinline__ unsigned int f2bf(float f) {
    unsigned int u = __builtin_bit_cast(unsigned int, f);
    return (u + 0x7FFFu + ((u >> 16) & 1u)) >> 16;
}
__device__ __forceinline__ unsigned int pkbf(float lo, float hi) {
    return f2bf(lo) | (f2bf(hi) << 16);
}

// fast round-to-nearest pack for the hot loop:
//   +0x8000 on raw bits (rounding carry propagates correctly for normals),
//   then one v_perm_b32 splices the two high halves: 3 VALU per dword.
__device__ __forceinline__ unsigned int pkbf_rn(float lo, float hi) {
    unsigned int ul = __builtin_bit_cast(unsigned int, lo) + 0x8000u;
    unsigned int uh = __builtin_bit_cast(unsigned int, hi) + 0x8000u;
    return __builtin_amdgcn_perm(uh, ul, 0x07060302u);  // {uh.b3,uh.b2,ul.b3,ul.b2}
}

// ---------------------------------------------------------------
// Pack W1 [768,384] f32 -> bf16 B-fragment layout:
//   frag(nt,kt): lane holds col n = nt*16+lane%16,
//   lanes 0-15 K=kt*32+{0..7,16..23}, lanes 16-31 K=kt*32+{8..15,24..31}, 2 bf16 per dword.
// ---------------------------------------------------------------
__global__ void pack_w1(const float* __restrict__ W1, unsigned int* __restrict__ out) {
    const int frag = blockIdx.x;               // nt*KT1 + kt
    const int nt = frag / KT1, kt = frag % KT1;
    const int lane = threadIdx.x;
    const int n = nt * 16 + (lane & 15);
    const int kbase = kt * 32 + ((lane & 16) ? 8 : 0);
    unsigned int* dst = out + ((size_t)frag * 32 + lane) * 8;
    #pragma unroll
    for (int i = 0; i < 8; ++i) {
        const int k0 = kbase + 2 * i + ((i >= 4) ? 8 : 0);
        dst[i] = pkbf(W1[(size_t)k0 * HID + n], W1[(size_t)(k0 + 1) * HID + n]);
    }
}

// Pack W2 [384,9] f32 -> bf16 B-fragment layout with N padded 9->16 (zeros).
__global__ void pack_w2(const float* __restrict__ W2, unsigned int* __restrict__ out) {
    const int kt = blockIdx.x;
    const int lane = threadIdx.x;
    const int n = lane & 15;
    const int kbase = kt * 32 + ((lane & 16) ? 8 : 0);
    unsigned int* dst = out + ((size_t)kt * 32 + lane) * 8;
    #pragma unroll
    for (int i = 0; i < 8; ++i) {
        const int k0 = kbase + 2 * i + ((i >= 4) ? 8 : 0);
        dst[i] = (n < NLAB)
                 ? pkbf(W2[(size_t)k0 * NLAB + n], W2[(size_t)(k0 + 1) * NLAB + n])
                 : 0u;
    }
}

// ---------------------------------------------------------------
// Fused MLP: h = tanh(X@W1+b1) (bf16 WMMA, f32 acc), staged in LDS,
// then logits = h@W2+b2 padded to 16 labels (cols 9..15 = -1000).
// One block = one 16-row M tile; 8 waves; wave w owns n-tiles 3w..3w+2.
// ---------------------------------------------------------------
__global__ __launch_bounds__(256)
void mlp_logits(const float* __restrict__ X,
                const float* __restrict__ b1v,
                const float* __restrict__ b2v,
                const unsigned int* __restrict__ w1p,
                const unsigned int* __restrict__ w2p,
                float* __restrict__ logits) {
    __shared__ unsigned short hT[16 * HID];    // 12 KB bf16 h tile
    const int mtile = blockIdx.x;
    const int lane  = threadIdx.x & 31;
    const int wave  = threadIdx.x >> 5;
    const int mrow  = lane & 15;               // A-frag row within tile
    const int half  = (lane & 16) ? 1 : 0;     // lane-group

    v8f acc0 = {}, acc1 = {}, acc2 = {};
    Frag16 a, b;

    const size_t rowBase = ((size_t)mtile * 16 + mrow) * D_IN;
    const int aoff = half ? 8 : 0;             // A layout: lanes16-31 start at K=8

    for (int kt = 0; kt < KT1; ++kt) {
        // A fragment: two contiguous 8-float runs, fast-packed f32 -> bf16
        const float* ap = X + rowBase + kt * 32 + aoff;
        v4f f0 = *(const v4f*)(ap);
        v4f f1 = *(const v4f*)(ap + 4);
        v4f f2 = *(const v4f*)(ap + 16);
        v4f f3 = *(const v4f*)(ap + 20);
        __builtin_prefetch(ap + 32, 0, 3);     // next k-tile -> global_prefetch_b8
        a.u[0] = pkbf_rn(f0[0], f0[1]); a.u[1] = pkbf_rn(f0[2], f0[3]);
        a.u[2] = pkbf_rn(f1[0], f1[1]); a.u[3] = pkbf_rn(f1[2], f1[3]);
        a.u[4] = pkbf_rn(f2[0], f2[1]); a.u[5] = pkbf_rn(f2[2], f2[3]);
        a.u[6] = pkbf_rn(f3[0], f3[1]); a.u[7] = pkbf_rn(f3[2], f3[3]);

        #pragma unroll
        for (int t = 0; t < 3; ++t) {
            const int nt = wave * 3 + t;
            const v4u* bp = (const v4u*)(w1p + ((size_t)(nt * KT1 + kt) * 32 + lane) * 8);
            b.q[0] = bp[0];
            b.q[1] = bp[1];
            v8f c = (t == 0) ? acc0 : (t == 1) ? acc1 : acc2;
            c = __builtin_amdgcn_wmma_f32_16x16x32_bf16(
                    false, a.v, false, b.v, (short)0, c, false, false);
            if (t == 0) acc0 = c; else if (t == 1) acc1 = c; else acc2 = c;
        }
    }

    // GEMM1 epilogue: + b1, tanh, stage bf16 h in LDS (C layout: col=lane%16, rows r+8*half)
    {
        const int ncol = lane & 15;
        #pragma unroll
        for (int t = 0; t < 3; ++t) {
            const int n = (wave * 3 + t) * 16 + ncol;
            const float bias = b1v[n];
            v8f c = (t == 0) ? acc0 : (t == 1) ? acc1 : acc2;
            #pragma unroll
            for (int r = 0; r < 8; ++r) {
                const float h = tanhf(c[r] + bias);
                hT[(r + half * 8) * HID + n] = (unsigned short)f2bf(h);
            }
        }
    }
    __syncthreads();

    // GEMM2 on wave 0: logits tile 16x16 over K=384 (12 WMMAs), A-frags from LDS
    if (wave == 0) {
        v8f c2 = {};
        Frag16 a2, b2f;
        for (int kt = 0; kt < KT2; ++kt) {
            const int base = mrow * HID + kt * 32 + aoff;     // 16B-aligned
            a2.q[0] = *(const v4u*)&hT[base];
            a2.q[1] = *(const v4u*)&hT[base + 16];
            const v4u* bp = (const v4u*)(w2p + ((size_t)kt * 32 + lane) * 8);
            b2f.q[0] = bp[0];
            b2f.q[1] = bp[1];
            c2 = __builtin_amdgcn_wmma_f32_16x16x32_bf16(
                    false, a2.v, false, b2f.v, (short)0, c2, false, false);
        }
        const int j = lane & 15;
        const float bias = (j < NLAB) ? b2v[j] : 0.0f;
        #pragma unroll
        for (int r = 0; r < 8; ++r) {
            const size_t row = (size_t)mtile * 16 + r + half * 8;
            const float val = (j < NLAB) ? (c2[r] + bias) : PAD_VAL; // pad_logits
            logits[row * LPAD + j] = val;
        }
    }
}

// ---------------------------------------------------------------
// Viterbi decode: one wave per batch element. Lane c owns label c.
// vit state broadcast across lanes via shuffles (no LDS sync needed).
// ---------------------------------------------------------------
__global__ __launch_bounds__(32)
void viterbi(const int* __restrict__ mask,
             const float* __restrict__ T,        // [NL,NL] row-major: T[cur*NL+prev]
             const float* __restrict__ logits,   // [MROWS,LPAD]
             unsigned char* __restrict__ ptrBuf, // [BATCH,SEQ,LPAD]
             int* __restrict__ preds) {          // [BATCH,SEQ]
    const int bIdx = blockIdx.x;
    const int lane = threadIdx.x;

    // lens[b] = sum(labels_mask[b])
    int cnt = 0;
    const int* mrow = mask + bIdx * SEQ;
    for (int i = lane; i < SEQ; i += 32) cnt += mrow[i];
    for (int m = 16; m >= 1; m >>= 1) cnt += __shfl_xor(cnt, m, 32);

    float Trow[NL];
    #pragma unroll
    for (int p = 0; p < NL; ++p) Trow[p] = (lane < NL) ? T[lane * NL + p] : 0.0f;
    const float Tend = (lane < NL) ? T[(NL - 1) * NL + lane] : 0.0f;  // transition[END][c]

    float vit = (lane == NL - 2) ? 0.0f : INIT_VAL;   // START = L-2
    const float* lg = logits + (size_t)bIdx * SEQ * LPAD;
    unsigned char* pw = ptrBuf + (size_t)bIdx * SEQ * LPAD;

    for (int s = 0; s < SEQ; ++s) {
        float best = -3.0e38f;
        int bp = 0;
        #pragma unroll
        for (int p = 0; p < NL; ++p) {                // ascending -> first-max (jnp.argmax)
            const float sc = __shfl(vit, p, 32) + Trow[p];
            if (sc > best) { best = sc; bp = p; }
        }
        if (lane < NL) pw[s * LPAD + lane] = (unsigned char)bp;
        const float lv = (lane < NL) ? lg[s * LPAD + lane] : PAD_VAL;
        float vout = (cnt > 0) ? (best + lv) : vit;
        if (cnt == 1) vout += Tend;                   // last real token: + T[END]
        vit = vout;
        --cnt;
    }

    // argmax over labels (first occurrence), then lane 0 backtracks
    int idx = 0;
    float bv = -3.0e38f;
    #pragma unroll
    for (int c = 0; c < NL; ++c) {
        const float v = __shfl(vit, c, 32);
        if (v > bv) { bv = v; idx = c; }
    }
    if (lane == 0) {
        int* op = preds + bIdx * SEQ;
        int cur = idx;
        for (int s = SEQ - 1; s >= 0; --s) {          // emit idx at s, then follow pointer
            op[s] = cur;
            cur = (int)pw[s * LPAD + cur];
        }
    }
}

// ---------------------------------------------------------------
extern "C" void kernel_launch(void* const* d_in, const int* in_sizes, int n_in,
                              void* d_out, int out_size, void* d_ws, size_t ws_size,
                              hipStream_t stream) {
    (void)in_sizes; (void)n_in; (void)out_size; (void)ws_size;
    const float* X    = (const float*)d_in[0];   // inputs  [B,S,D]
    const int*   mask = (const int*)  d_in[1];   // labels_mask [B,S] int32
    const float* W1   = (const float*)d_in[2];   // [D,HID]
    const float* b1v  = (const float*)d_in[3];   // [HID]
    const float* W2   = (const float*)d_in[4];   // [HID,NLAB]
    const float* b2v  = (const float*)d_in[5];   // [NLAB]
    const float* T    = (const float*)d_in[6];   // [NL,NL]

    char* ws = (char*)d_ws;
    unsigned int*  w1p    = (unsigned int*)(ws + WS_W1_OFF);
    unsigned int*  w2p    = (unsigned int*)(ws + WS_W2_OFF);
    float*         logits = (float*)(ws + WS_LG_OFF);
    unsigned char* ptrBuf = (unsigned char*)(ws + WS_PTR_OFF);
    int*           preds  = (int*)d_out;

    pack_w1<<<NT1 * KT1, 32, 0, stream>>>(W1, w1p);
    pack_w2<<<KT2, 32, 0, stream>>>(W2, w2p);
    mlp_logits<<<MROWS / 16, 256, 0, stream>>>(X, b1v, b2v, w1p, w2p, logits);
    viterbi<<<BATCH, 32, 0, stream>>>(mask, T, logits, ptrBuf, preds);
}
